// TriangleMultiplicativeUpdate_30588757082756
// MI455X (gfx1250) — matrix operand order repaired and checked by reading, not compiled, not run on hardware
//
#include <hip/hip_runtime.h>
#include <hip/hip_bf16.h>
#include <math.h>

// ---------------------------------------------------------------------------
// TriangleMultiplicativeUpdate (outgoing), B=1, N=768, c_z=c_h=128, fp32 in/out.
// Strategy (MI455X / gfx1250, wave32, WMMA):
//   * f32 LayerNorms, f16 activations, v_wmma_f32_16x16x32_f16 for all GEMMs
//     (f32 accumulate).
//   * a,b,p all kept channel-major [c][m] so both the einsum operands and its
//     result stores are contiguous (no 4B-scatter write amplification).
//   * Triangle einsum: block-per-channel 128x128 tile, async global->LDS
//     staging (GLOBAL_LOAD_ASYNC_TO_LDS_B128 + s_wait_asynccnt), double
//     buffered, 8 waves x 8 WMMA tiles each -> 64 flop/byte from global.
// ---------------------------------------------------------------------------

typedef _Float16 v8h  __attribute__((ext_vector_type(8)));
typedef _Float16 v16h __attribute__((ext_vector_type(16)));
typedef float    v8f  __attribute__((ext_vector_type(8)));

#define NPOS 768
#define CDIM 128
#define MTOT ((size_t)NPOS * NPOS)   // 589824 rows

__device__ __forceinline__ float sigf(float x) { return 1.0f / (1.0f + __expf(-x)); }

// CDNA5 async copy: LDS[lds_off] = global[ga], 16B per lane, ASYNCcnt-tracked.
__device__ __forceinline__ void async_ld_b128(unsigned lds_off, unsigned long long ga) {
    asm volatile("global_load_async_to_lds_b128 %0, %1, off"
                 :: "v"(lds_off), "v"(ga) : "memory");
}
__device__ __forceinline__ void wait_async0() {
    asm volatile("s_wait_asynccnt 0x0" ::: "memory");
}

// Build v16h A-fragment per CDNA5 16-bit A 16x32 layout (CDNA5 ISA):
//   lane L: M = L%16, hi = L/16; elements 0..7 hold K = 8*hi + 0..7,
//   elements 8..15 hold K = 16 + 8*hi + 0..7.
__device__ __forceinline__ v16h make_afrag(const _Float16* row, int kk, int hi) {
    v8h lo = *(const v8h*)(row + kk + 8 * hi);
    v8h hi8 = *(const v8h*)(row + kk + 16 + 8 * hi);
    v16h f;
#pragma unroll
    for (int e = 0; e < 8; ++e) { f[e] = lo[e]; f[e + 8] = hi8[e]; }
    return f;
}

// B-fragment from a row-major B^T [N][K] buffer: lane L: N = L%16, hi = L/16;
// elements 0..7 hold K = 8*hi + 0..7, elements 8..15 hold K = 16 + 8*hi + 0..7.
__device__ __forceinline__ v16h make_bfrag(const _Float16* rowT, int kk, int hi) {
    v8h lo = *(const v8h*)(rowT + kk + 8 * hi);
    v8h hi8 = *(const v8h*)(rowT + kk + 16 + 8 * hi);
    v16h f;
#pragma unroll
    for (int e = 0; e < 8; ++e) { f[e] = lo[e]; f[e + 8] = hi8[e]; }
    return f;
}

// ---------------------------------------------------------------------------
// k0: convert 6 weight matrices (128x128, row-major [out][in]) to f16.
// [out][in] row-major == exactly the B^T [N][K] layout WMMA consumes.
// Order: a_g, a_p, b_g, b_p, g_w, z_w.
// ---------------------------------------------------------------------------
__global__ __launch_bounds__(256) void k0_cvt_weights(
    const float* __restrict__ ag, const float* __restrict__ ap,
    const float* __restrict__ bg, const float* __restrict__ bp,
    const float* __restrict__ gw, const float* __restrict__ zw,
    _Float16* __restrict__ dst) {
    int i = blockIdx.x * 256 + threadIdx.x;        // 0 .. 6*16384-1
    int mat = i >> 14;
    int idx = i & 16383;
    const float* s;
    switch (mat) {
        case 0: s = ag; break; case 1: s = ap; break; case 2: s = bg; break;
        case 3: s = bp; break; case 4: s = gw; break; default: s = zw; break;
    }
    dst[i] = (_Float16)s[idx];
}

// ---------------------------------------------------------------------------
// k1: per 16-row tile of the flattened [M=589824, 128] problem:
//   LayerNorm(z) -> f16 LDS tile -> 5 GEMMs (a_g,a_p,b_g,b_p,g) via WMMA ->
//   fused sigmoid/mask gating -> a,b f16 channel-major [c][m], g f16 [m][c].
// ---------------------------------------------------------------------------
__global__ __launch_bounds__(256) void k1_ln_proj(
    const float* __restrict__ z, const float* __restrict__ mask,
    const float* __restrict__ nw, const float* __restrict__ nb,
    const _Float16* __restrict__ wts,  // [5][128][128] f16
    _Float16* __restrict__ a_h, _Float16* __restrict__ b_h,
    _Float16* __restrict__ g_h) {
    __shared__ _Float16 lnT[16][136];     // padded pitch: 272B -> bank-rotating
    __shared__ float    cbuf[5][16][132]; // raw GEMM results, padded pitch

    const int mtile = blockIdx.x * 16;
    const int t = threadIdx.x;

    // ---- Phase A: LayerNorm of 16 rows (16 lanes per row, 8 vals per lane)
    {
        int r = t >> 4, l16 = t & 15;
        size_t m = (size_t)(mtile + r);
        const float* zr = z + m * CDIM + l16 * 8;
        float v[8];
        float s = 0.f, s2 = 0.f;
#pragma unroll
        for (int e = 0; e < 8; ++e) { v[e] = zr[e]; s += v[e]; s2 += v[e] * v[e]; }
#pragma unroll
        for (int off = 8; off; off >>= 1) {
            s  += __shfl_xor(s,  off, 16);
            s2 += __shfl_xor(s2, off, 16);
        }
        float mu = s * (1.0f / 128.0f);
        float var = s2 * (1.0f / 128.0f) - mu * mu;
        float rs = rsqrtf(var + 1e-5f);
#pragma unroll
        for (int e = 0; e < 8; ++e) {
            int c = l16 * 8 + e;
            lnT[r][c] = (_Float16)(((v[e] - mu) * rs) * nw[c] + nb[c]);
        }
    }
    __syncthreads();

    // ---- Phase B: 40 WMMA tasks (5 matrices x 8 N-tiles) over 8 waves
    {
        int wave = t >> 5, lane = t & 31;
        int l16 = lane & 15, hi = lane >> 4;
        for (int task = wave; task < 40; task += 8) {
            int mat = task >> 3, nt = task & 7;
            const _Float16* wrow = wts + (size_t)mat * 16384 + (size_t)(nt * 16 + l16) * 128;
            const _Float16* arow = &lnT[l16][0];
            v8f acc = {};
#pragma unroll
            for (int kk = 0; kk < 128; kk += 32) {
                v16h af = make_afrag(arow, kk, hi);
                v16h bf = make_bfrag(wrow, kk, hi);
                acc = __builtin_amdgcn_wmma_f32_16x16x32_f16(
                    false, af, false, bf, (short)0, acc, false, false);
            }
#pragma unroll
            for (int v = 0; v < 8; ++v) cbuf[mat][v + 8 * hi][nt * 16 + l16] = acc[v];
        }
    }
    __syncthreads();

    // ---- Phase C1: a,b -> f16 channel-major [c][m] (packed 16B stores)
    {
        int ch = t >> 1, half = t & 1, r0 = half * 8;
        v8h av, bv;
#pragma unroll
        for (int e = 0; e < 8; ++e) {
            int r = r0 + e;
            float mk = mask[mtile + r];
            av[e] = (_Float16)(sigf(cbuf[0][r][ch]) * cbuf[1][r][ch] * mk);
            bv[e] = (_Float16)(sigf(cbuf[2][r][ch]) * cbuf[3][r][ch] * mk);
        }
        *(v8h*)(a_h + (size_t)ch * MTOT + mtile + r0) = av;
        *(v8h*)(b_h + (size_t)ch * MTOT + mtile + r0) = bv;
    }
    // ---- Phase C2: g = sigmoid(.) -> f16 row-major [m][c]
    {
        int r = t >> 4, c8 = (t & 15) * 8;
        v8h gv;
#pragma unroll
        for (int e = 0; e < 8; ++e) gv[e] = (_Float16)sigf(cbuf[4][r][c8 + e]);
        *(v8h*)(g_h + (size_t)(mtile + r) * CDIM + c8) = gv;
    }
}

// ---------------------------------------------------------------------------
// k2: triangle einsum  p[c][i*768+j] = sum_k a[c][i*768+k] * b[c][j*768+k]
// grid (6 j-blocks, 6 i-blocks, 128 channels); block = 8 waves computes one
// channel's 128x128 (i,j) tile. Per 32-K chunk: async-stage A[128x32] and
// B[128x32] f16 into LDS (double buffered, 80B padded row pitch), then each
// wave (2x4 wave grid, 4x2 accumulator tiles) does 12 ds_load_b128 + 8 WMMA.
// p is stored channel-major -> 64B-contiguous stores per accumulator VGPR.
// ---------------------------------------------------------------------------
__global__ __launch_bounds__(256) void k2_triangle(
    const _Float16* __restrict__ a_h, const _Float16* __restrict__ b_h,
    float* __restrict__ p) {
    // [sel][A|B]: A at +0, B at +10240; sel stride 20480. 128 rows x 80B.
    __shared__ char smem[40960];

    const int t = threadIdx.x;
    const int w = t >> 5, lane = t & 31;
    const int l16 = lane & 15, hi = lane >> 4;
    const int it128 = blockIdx.y * 128, jt128 = blockIdx.x * 128;
    const int c = blockIdx.z;
    const int R = w >> 2, C = w & 3;                 // 2x4 wave grid

    const _Float16* Ab = a_h + (size_t)c * MTOT + (size_t)it128 * NPOS;
    const _Float16* Bb = b_h + (size_t)c * MTOT + (size_t)jt128 * NPOS;

    const unsigned smbase = (unsigned)(size_t)(&smem[0]);
    const int rr = w * 16 + (lane >> 2);             // staging row (sub 0), <128
    const int q  = lane & 3;                         // 16B quarter of 64B row

    // Issue the async copies for one 32-K chunk: each wave moves 2x8 rows of A
    // and of B (4 instructions, 512B each).
    auto issue = [&](int kk, int sel) {
        unsigned ldsA = smbase + (unsigned)sel * 20480u + (unsigned)rr * 80u + (unsigned)q * 16u;
        unsigned long long gA = (unsigned long long)(Ab + (size_t)rr * NPOS + kk) + (unsigned long long)q * 16u;
        unsigned long long gB = (unsigned long long)(Bb + (size_t)rr * NPOS + kk) + (unsigned long long)q * 16u;
        async_ld_b128(ldsA,              gA);
        async_ld_b128(ldsA + 8u * 80u,   gA + 8ull * NPOS * 2ull);
        async_ld_b128(ldsA + 10240u,     gB);
        async_ld_b128(ldsA + 10240u + 8u * 80u, gB + 8ull * NPOS * 2ull);
    };

    v8f acc[4][2] = {};

    issue(0, 0);
    wait_async0();
    __syncthreads();

    for (int ch = 0; ch < NPOS / 32; ++ch) {
        const int sel = ch & 1;
        if (ch + 1 < NPOS / 32) issue((ch + 1) * 32, sel ^ 1);

        const char* Al = smem + sel * 20480;
        const char* Bl = Al + 10240;
        v16h af[4], bf[2];
#pragma unroll
        for (int mi = 0; mi < 4; ++mi) {
            const char* base = Al + ((4 * R + mi) * 16 + l16) * 80;
            v8h lo  = *(const v8h*)(base + hi * 16);
            v8h hi8 = *(const v8h*)(base + 32 + hi * 16);
#pragma unroll
            for (int e = 0; e < 8; ++e) { af[mi][e] = lo[e]; af[mi][e + 8] = hi8[e]; }
        }
#pragma unroll
        for (int nj = 0; nj < 2; ++nj) {
            const char* base = Bl + ((2 * C + nj) * 16 + l16) * 80;
            v8h lo  = *(const v8h*)(base + hi * 16);
            v8h hi8 = *(const v8h*)(base + 32 + hi * 16);
#pragma unroll
            for (int e = 0; e < 8; ++e) { bf[nj][e] = lo[e]; bf[nj][e + 8] = hi8[e]; }
        }
#pragma unroll
        for (int mi = 0; mi < 4; ++mi)
#pragma unroll
            for (int nj = 0; nj < 2; ++nj)
                acc[mi][nj] = __builtin_amdgcn_wmma_f32_16x16x32_f16(
                    false, af[mi], false, bf[nj], (short)0, acc[mi][nj], false, false);

        wait_async0();       // own staging for next chunk done
        __syncthreads();     // everyone's staging done / this buffer consumed
    }

    // C layout: lane L, vgpr v -> row = v + 8*hi, col = l16.
    // p channel-major: 16 lanes -> 64B contiguous per accumulator VGPR.
    float* pc = p + (size_t)c * MTOT;
#pragma unroll
    for (int mi = 0; mi < 4; ++mi) {
#pragma unroll
        for (int nj = 0; nj < 2; ++nj) {
            size_t i0 = (size_t)(it128 + (4 * R + mi) * 16 + 8 * hi);
            size_t j  = (size_t)(jt128 + (2 * C + nj) * 16 + l16);
#pragma unroll
            for (int v = 0; v < 8; ++v)
                pc[(i0 + v) * NPOS + j] = acc[mi][nj][v];
        }
    }
}

// ---------------------------------------------------------------------------
// k3: out = sigmoid_gate * (LayerNorm(p) @ z_w^T), f32 output.
// p is channel-major [c][m]; stage the 128(c) x 16(m) patch through LDS with
// coalesced 32B-per-thread global loads, then LayerNorm rows from LDS.
// ---------------------------------------------------------------------------
__global__ __launch_bounds__(256) void k3_out(
    const float* __restrict__ p, const _Float16* __restrict__ g_h,
    const _Float16* __restrict__ zw_h, const float* __restrict__ now,
    const float* __restrict__ nob, float* __restrict__ out) {
    __shared__ float    pT[16][132];      // [m-row][channel], padded pitch
    __shared__ _Float16 lnT[16][136];
    const int mtile = blockIdx.x * 16;
    const int t = threadIdx.x;

    {   // Phase A0: coalesced gather of the channel-major patch + transpose
        int c = t >> 1, half = t & 1;
        const float* src = p + (size_t)c * MTOT + mtile + half * 8;
#pragma unroll
        for (int e = 0; e < 8; ++e) pT[half * 8 + e][c] = src[e];
    }
    __syncthreads();

    {   // Phase A: LayerNorm over channels (rows now in LDS)
        int r = t >> 4, l16 = t & 15;
        const float* pr = &pT[r][l16 * 8];
        float v[8];
        float s = 0.f, s2 = 0.f;
#pragma unroll
        for (int e = 0; e < 8; ++e) { v[e] = pr[e]; s += v[e]; s2 += v[e] * v[e]; }
#pragma unroll
        for (int off = 8; off; off >>= 1) {
            s  += __shfl_xor(s,  off, 16);
            s2 += __shfl_xor(s2, off, 16);
        }
        float mu = s * (1.0f / 128.0f);
        float var = s2 * (1.0f / 128.0f) - mu * mu;
        float rs = rsqrtf(var + 1e-5f);
#pragma unroll
        for (int e = 0; e < 8; ++e) {
            int cc = l16 * 8 + e;
            lnT[r][cc] = (_Float16)(((v[e] - mu) * rs) * now[cc] + nob[cc]);
        }
    }
    __syncthreads();

    int wave = t >> 5, lane = t & 31;
    int l16 = lane & 15, hi = lane >> 4;
    int nt = wave;                                   // 8 waves == 8 N-tiles
    const _Float16* wrow = zw_h + (size_t)(nt * 16 + l16) * 128;
    const _Float16* arow = &lnT[l16][0];
    v8f acc = {};
#pragma unroll
    for (int kk = 0; kk < 128; kk += 32) {
        v16h af = make_afrag(arow, kk, hi);
        v16h bf = make_bfrag(wrow, kk, hi);
        acc = __builtin_amdgcn_wmma_f32_16x16x32_f16(
            false, af, false, bf, (short)0, acc, false, false);
    }
    int col = nt * 16 + l16;
#pragma unroll
    for (int v = 0; v < 8; ++v) {
        size_t m = (size_t)(mtile + v + 8 * hi);
        float gg = (float)g_h[m * CDIM + col];
        out[m * CDIM + col] = gg * acc[v];           // 16 lanes -> 64B contiguous
    }
}

// ---------------------------------------------------------------------------
extern "C" void kernel_launch(void* const* d_in, const int* in_sizes, int n_in,
                              void* d_out, int out_size, void* d_ws, size_t ws_size,
                              hipStream_t stream) {
    const float* z    = (const float*)d_in[0];
    const float* mask = (const float*)d_in[1];
    const float* niw  = (const float*)d_in[2];
    const float* nib  = (const float*)d_in[3];
    const float* agw  = (const float*)d_in[4];
    const float* apw  = (const float*)d_in[5];
    const float* bgw  = (const float*)d_in[6];
    const float* bpw  = (const float*)d_in[7];
    const float* now  = (const float*)d_in[8];
    const float* nob  = (const float*)d_in[9];
    const float* gw   = (const float*)d_in[10];
    const float* zw   = (const float*)d_in[11];
    float* out = (float*)d_out;

    // Workspace carve-up (all 256B-aligned offsets):
    //   wts : 6*16384 f16 = 196,608 B
    //   a_h, b_h, g_h : MTOT*128 f16 = 150,994,944 B each
    //   p   : MTOT*128 f32 = 301,989,888 B (channel-major [c][m])
    char* ws = (char*)d_ws;
    const size_t MC = MTOT * (size_t)CDIM;
    _Float16* wts = (_Float16*)ws;
    _Float16* a_h = (_Float16*)(ws + 196608);
    _Float16* b_h = (_Float16*)(ws + 196608 + MC * 2);
    _Float16* g_h = (_Float16*)(ws + 196608 + MC * 4);
    float*    pbf = (float*)   (ws + 196608 + MC * 6);
    _Float16* zw_h = wts + 5 * 16384;

    const int mtiles = (int)(MTOT / 16);   // 36864

    k0_cvt_weights<<<dim3(384), dim3(256), 0, stream>>>(agw, apw, bgw, bpw, gw, zw, wts);
    k1_ln_proj<<<dim3(mtiles), dim3(256), 0, stream>>>(z, mask, niw, nib, wts,
                                                       a_h, b_h, g_h);
    k2_triangle<<<dim3(NPOS / 128, NPOS / 128, CDIM), dim3(256), 0, stream>>>(
        a_h, b_h, pbf);
    k3_out<<<dim3(mtiles), dim3(256), 0, stream>>>(pbf, g_h, zw_h, now, nob, out);
}
